// DGLJTNNDecoder_69002944577977
// MI455X (gfx1250) — matrix operand order repaired and checked by reading, not compiled, not run on hardware
//
#include <hip/hip_runtime.h>
#include <hip/hip_bf16.h>
#include <math.h>

// ---------------- problem constants (from reference) ----------------
#define BB 512            // batch of trees
#define NT 20             // nodes per tree
#define HD 450            // hidden
#define LD 56             // tree_vec dim
#define VV 780            // vocab
#define TT 38             // 2*(NT-1) DFS steps
#define NNODES (BB*NT)    // 10240
#define NEDGES (BB*2*(NT-1)) // 19456
#define NPROWS ((TT+1)*BB)   // 19968 p rows (multiple of 64)

// padded dims (K mult of 32, N mult of 128, zero-filled pads)
#define KA   928          // 2H = 900 -> 928   (recurrent GEMMs)
#define KPIN 960          // 2H+L = 956 -> 960 (p head)
#define KQIN 512          // H+L  = 506 -> 512 (q head)
#define KH2  512          // H    = 450 -> 512 (q logits GEMM K == q_hid ld)
#define NPH  512          // padded N for H(450)-column outputs
#define NPV  896          // padded N for V(780)-column outputs

// GEMM tiling
#define TBM 64
#define TBN 128
#define TBK 32
#define AST 40            // LDS row stride in halves: 80B (16B-aligned rows, 16 distinct banks)

typedef __attribute__((ext_vector_type(16))) _Float16 v16h;
typedef __attribute__((ext_vector_type(8)))  float    v8f;
typedef __attribute__((ext_vector_type(4)))  int      v4i;

// -------- optional CDNA5 async global->LDS staging (compile-time probed) ----
#if __has_builtin(__builtin_amdgcn_global_load_async_to_lds_b128) && \
    __has_builtin(__builtin_amdgcn_s_wait_asynccnt)
#define HAVE_ASYNC_LDS 1
typedef v4i __attribute__((address_space(1))) * gv4_t;   // global v4i*
typedef v4i __attribute__((address_space(3))) * lv4_t;   // LDS v4i*
#else
#define HAVE_ASYNC_LDS 0
#endif

// ---------------- WMMA fragment loads (CDNA5 16-bit layouts, ISA 7.12.2) ----
__device__ inline v16h frag_a(const _Float16* __restrict__ As, int tileRow, int lane) {
  // A 16x32 f16: lane m = lane&15; lanes<16 hold K 0..7,16..23; lanes>=16 hold K 8..15,24..31
  const _Float16* p = As + (size_t)(tileRow + (lane & 15)) * AST;
  const int kb = (lane & 16) ? 8 : 0;
  v16h f;
#pragma unroll
  for (int j = 0; j < 8; ++j) f[j] = p[kb + j];
#pragma unroll
  for (int j = 0; j < 8; ++j) f[8 + j] = p[kb + 16 + j];
  return f;
}
__device__ inline v16h frag_b(const _Float16* __restrict__ Bs, int tileCol, int lane) {
  // Bs holds the B tile transposed: Bs[n][k]; lane n = lane&15; lanes<16 K 0..7,16..23; lanes>=16 K 8..15,24..31
  const _Float16* p = Bs + (size_t)(tileCol + (lane & 15)) * AST + ((lane & 16) ? 8 : 0);
  v16h f;
#pragma unroll
  for (int j = 0; j < 16; ++j) f[j] = p[(j < 8) ? j : j + 8];
  return f;
}

// ---------------- guard-free f16 WMMA GEMM -------------------------------
//   C[M,NP] = act(A[M,K] @ W[K,NP] + biasP[NP])
// Requirements (enforced by callers): M%64==0, NP%128==0, K%32==0,
// lda==K, ldb==ldc==NP, W rows K..Kpad and cols N..NP zero-filled,
// biasP zero-padded.  ACT: 0 none, 1 sigmoid, 2 tanh, 3 relu.
template <int ACT, int HALF_OUT>
__global__ __launch_bounds__(256) void k_gemm(
    const _Float16* __restrict__ A, const _Float16* __restrict__ W,
    const float* __restrict__ biasP, void* __restrict__ Cout,
    int NP, int K) {
  __shared__ alignas(16) _Float16 As[TBM * AST];
  __shared__ alignas(16) _Float16 Bs[TBN * AST];
  const int tid  = threadIdx.x;
  const int lane = tid & 31;
  const int wave = tid >> 5;                 // 8 waves: 2 (M) x 4 (N)
  const int wm = wave >> 2, wn = wave & 3;
  const int bm = blockIdx.y * TBM, bn = blockIdx.x * TBN;

  // A staging: thread copies 16B of the 64x32 tile
  const int ar  = tid >> 2;                  // 0..63
  const int ac0 = (tid & 3) * 8;             // 0,8,16,24
  // B staging: thread handles 2 K-rows x 8 N-cols of the 32x128 tile
  const int kp  = tid >> 4;                  // 0..15 -> k = 2kp, 2kp+1
  const int n0  = (tid & 15) * 8;            // 0..120

  v8f acc[2][2] = {};

  for (int kt = 0; kt < K; kt += TBK) {
    // ---- stage A tile (async DMA to LDS when available) ----
    {
      const _Float16* src = A + (size_t)(bm + ar) * K + kt + ac0;
      _Float16* dst = As + (size_t)ar * AST + ac0;
#if HAVE_ASYNC_LDS
      __builtin_amdgcn_global_load_async_to_lds_b128((gv4_t)src, (lv4_t)dst, 0, 0);
#else
      *(int4*)dst = *(const int4*)src;
#endif
    }
    // ---- stage B tile transposed into Bs[n][k] (packed k-pair stores) ----
    {
      const _Float16* src0 = W + (size_t)(kt + 2 * kp) * NP + bn + n0;
      if (kt + TBK < K)  // prefetch next K tile of W (global_prefetch_b8)
        __builtin_prefetch(src0 + (size_t)TBK * NP, 0, 1);
      unsigned short ra[8], rb[8];
      *(uint4*)ra = *(const uint4*)src0;
      *(uint4*)rb = *(const uint4*)(src0 + NP);
#pragma unroll
      for (int j = 0; j < 8; ++j) {
        const unsigned v = (unsigned)ra[j] | ((unsigned)rb[j] << 16);
        *(unsigned*)(Bs + (size_t)(n0 + j) * AST + 2 * kp) = v;
      }
    }
#if HAVE_ASYNC_LDS
    __builtin_amdgcn_s_wait_asynccnt(0);
#endif
    __syncthreads();

    v16h a0 = frag_a(As, wm * 32,      lane);
    v16h a1 = frag_a(As, wm * 32 + 16, lane);
    v16h b0 = frag_b(Bs, wn * 32,      lane);
    v16h b1 = frag_b(Bs, wn * 32 + 16, lane);
    acc[0][0] = __builtin_amdgcn_wmma_f32_16x16x32_f16(false, a0, false, b0, (short)0, acc[0][0], false, false);
    acc[0][1] = __builtin_amdgcn_wmma_f32_16x16x32_f16(false, a0, false, b1, (short)0, acc[0][1], false, false);
    acc[1][0] = __builtin_amdgcn_wmma_f32_16x16x32_f16(false, a1, false, b0, (short)0, acc[1][0], false, false);
    acc[1][1] = __builtin_amdgcn_wmma_f32_16x16x32_f16(false, a1, false, b1, (short)0, acc[1][1], false, false);
    __syncthreads();
  }

  // ---- epilogue (unguarded): D layout -> M = vgpr + 8*(lane>=16), N = lane&15
  const int mrow0 = bm + wm * 32;
  const int ncol0 = bn + wn * 32;
  const int mofs  = (lane & 16) ? 8 : 0;
  const int nofs  = lane & 15;
#pragma unroll
  for (int ti = 0; ti < 2; ++ti)
#pragma unroll
    for (int tj = 0; tj < 2; ++tj)
#pragma unroll
      for (int r = 0; r < 8; ++r) {
        const int m_ = mrow0 + ti * 16 + r + mofs;
        const int n_ = ncol0 + tj * 16 + nofs;
        float v = acc[ti][tj][r] + biasP[n_];
        if (ACT == 1) v = 1.f / (1.f + __expf(-v));
        else if (ACT == 2) v = tanhf(v);
        else if (ACT == 3) v = fmaxf(v, 0.f);
        if (HALF_OUT) ((_Float16*)Cout)[(size_t)m_ * NP + n_] = (_Float16)v;
        else          ((float*)Cout)[(size_t)m_ * NP + n_] = v;
      }
}

// ---------------- small utility kernels ----------------
__global__ void k_zero_u32(unsigned* __restrict__ p, size_t n) {
  size_t i = (size_t)blockIdx.x * blockDim.x + threadIdx.x;
  const size_t st = (size_t)gridDim.x * blockDim.x;
  for (; i < n; i += st) p[i] = 0u;
}

// f32 [K,Nc] -> f16 [KP,NPc] zero-filling row and column pads
__global__ void k_cvt(const float* __restrict__ src, _Float16* __restrict__ dst,
                      int K, int Nc, int KP, int NPc) {
  const size_t idx = (size_t)blockIdx.x * blockDim.x + threadIdx.x;
  const size_t tot = (size_t)KP * NPc;
  if (idx >= tot) return;
  const int k = (int)(idx / NPc), n = (int)(idx % NPc);
  dst[idx] = (k < K && n < Nc) ? (_Float16)src[(size_t)k * Nc + n] : (_Float16)0.f;
}

// f32 bias [Nc] -> f32 [NPc] zero-padded
__global__ void k_cvt_bias(const float* __restrict__ src, float* __restrict__ dst,
                           int Nc, int NPc) {
  const int i = blockIdx.x * blockDim.x + threadIdx.x;
  if (i < NPc) dst[i] = (i < Nc) ? src[i] : 0.f;
}

// x_h[node,h] = (f16) emb[wid[node], h]
__global__ void k_gather_emb(const float* __restrict__ emb, const int* __restrict__ wid,
                             _Float16* __restrict__ x_h) {
  const size_t idx = (size_t)blockIdx.x * blockDim.x + threadIdx.x;
  if (idx >= (size_t)NNODES * HD) return;
  const int node = (int)(idx / HD), h = (int)(idx % HD);
  x_h[idx] = (_Float16)emb[(size_t)wid[node] * HD + h];
}

// per-step: pred sums + build A1=[sx|s], A2=[sx|arm]   (S buffers ld NPH)
__global__ void k_step_build(const int* __restrict__ eid, const int* __restrict__ esrc,
                             const int* __restrict__ epred, int P,
                             const float* __restrict__ m, const float* __restrict__ rm,
                             const _Float16* __restrict__ x_h,
                             float* __restrict__ Sbuf,
                             _Float16* __restrict__ A1, _Float16* __restrict__ A2) {
  const int idx = blockIdx.x * blockDim.x + threadIdx.x;
  if (idx >= BB * HD) return;
  const int b = idx / HD, h = idx - b * HD;
  const int e = eid[b];
  float s = 0.f, ar = 0.f;
  for (int p = 0; p < P; ++p) {
    const int pe = epred[(size_t)e * P + p];       // sentinel NEDGES -> zero row
    s  += m [(size_t)pe * HD + h];
    ar += rm[(size_t)pe * HD + h];
  }
  const _Float16 sx = x_h[(size_t)esrc[e] * HD + h];
  A1[(size_t)b * KA + h]      = sx;
  A1[(size_t)b * KA + HD + h] = (_Float16)s;
  A2[(size_t)b * KA + h]      = sx;
  A2[(size_t)b * KA + HD + h] = (_Float16)ar;
  Sbuf[(size_t)b * NPH + h] = s;
}

// m_new = (1-z)*s + z*tanh_pre; build A3 = [dx | m_new]
__global__ void k_mnew(const int* __restrict__ eid, const int* __restrict__ edst,
                       const float* __restrict__ Z, const float* __restrict__ Hh,
                       const float* __restrict__ Sbuf, const _Float16* __restrict__ x_h,
                       float* __restrict__ Mnew, _Float16* __restrict__ A3) {
  const int idx = blockIdx.x * blockDim.x + threadIdx.x;
  if (idx >= BB * HD) return;
  const int b = idx / HD, h = idx - b * HD;
  const size_t o = (size_t)b * NPH + h;
  const float z = Z[o];
  const float mn = (1.f - z) * Sbuf[o] + z * Hh[o];
  Mnew[o] = mn;
  const int e = eid[b];
  A3[(size_t)b * KA + h]      = x_h[(size_t)edst[e] * HD + h];
  A3[(size_t)b * KA + HD + h] = (_Float16)mn;
}

// scatter m[eid] = m_new ; rm[eid] = r*m_new
__global__ void k_scatter(const int* __restrict__ eid, const float* __restrict__ Mnew,
                          const float* __restrict__ R,
                          float* __restrict__ m, float* __restrict__ rm) {
  const int idx = blockIdx.x * blockDim.x + threadIdx.x;
  if (idx >= BB * HD) return;
  const int b = idx / HD, h = idx - b * HD;
  const int e = eid[b];
  const float mn = Mnew[(size_t)b * NPH + h];
  m [(size_t)e * HD + h] = mn;
  rm[(size_t)e * HD + h] = R[(size_t)b * NPH + h] * mn;
}

// h = sum over node_in[v] of m rows
__global__ void k_pull(const int* __restrict__ sv, const int* __restrict__ node_in, int Dn,
                       const float* __restrict__ m, float* __restrict__ h_row_base) {
  const int idx = blockIdx.x * blockDim.x + threadIdx.x;
  if (idx >= BB * HD) return;
  const int b = idx / HD, h = idx - b * HD;
  const int v = sv[b];
  float acc = 0.f;
  for (int d = 0; d < Dn; ++d) {
    const int ee = node_in[(size_t)v * Dn + d];    // sentinel NEDGES -> zero row
    acc += m[(size_t)ee * HD + h];
  }
  h_row_base[idx] = acc;
}

// p_in row r: [ x[node] | h_all[r] | tree_vec[r%B] ]  (ld KPIN, pads pre-zeroed)
__global__ void k_build_pin(const _Float16* __restrict__ x_h, const float* __restrict__ h_all,
                            const float* __restrict__ tv, const int* __restrict__ root_ids,
                            const int* __restrict__ step_v, _Float16* __restrict__ p_in) {
  const size_t idx = (size_t)blockIdx.x * blockDim.x + threadIdx.x;
  const size_t tot = (size_t)NPROWS * (2 * HD + LD);
  if (idx >= tot) return;
  const int r = (int)(idx / (2 * HD + LD));
  const int c = (int)(idx % (2 * HD + LD));
  const int b = r % BB;
  _Float16 v;
  if (c < HD) {
    const int node = (r < BB) ? root_ids[b] : step_v[(size_t)(r / BB - 1) * BB + b];
    v = x_h[(size_t)node * HD + c];
  } else if (c < 2 * HD) {
    v = (_Float16)h_all[(size_t)r * HD + (c - HD)];
  } else {
    v = (_Float16)tv[(size_t)b * LD + (c - 2 * HD)];
  }
  p_in[(size_t)r * KPIN + c] = v;
}

// q_in row i (selected by q_rows): [ h_all[qr] | tree_vec[qr%B] ] (ld KQIN)
__global__ void k_build_qin(const float* __restrict__ h_all, const float* __restrict__ tv,
                            const int* __restrict__ q_rows, _Float16* __restrict__ q_in, int nq) {
  const size_t idx = (size_t)blockIdx.x * blockDim.x + threadIdx.x;
  const size_t tot = (size_t)nq * (HD + LD);
  if (idx >= tot) return;
  const int i = (int)(idx / (HD + LD));
  const int c = (int)(idx % (HD + LD));
  const int r = q_rows[i];
  _Float16 v;
  if (c < HD) v = (_Float16)h_all[(size_t)r * HD + c];
  else        v = (_Float16)tv[(size_t)(r % BB) * LD + (c - HD)];
  q_in[(size_t)i * KQIN + c] = v;
}

// p_logit[row] = dot(p_hid[row,:450], Us) + bs   (one wave per row; p_hid ld NPH)
__global__ void k_plogit(const _Float16* __restrict__ p_hid, const float* __restrict__ Us,
                         const float* __restrict__ bs, float* __restrict__ p_logit, int Rows) {
  const int row = blockIdx.x * (blockDim.x >> 5) + (threadIdx.x >> 5);
  const int lane = threadIdx.x & 31;
  if (row >= Rows) return;
  const _Float16* r = p_hid + (size_t)row * NPH;
  float acc = 0.f;
  for (int h = lane; h < HD; h += 32) acc += (float)r[h] * Us[h];
  for (int off = 16; off; off >>= 1) acc += __shfl_down(acc, off, 32);
  if (lane == 0) p_logit[row] = acc + bs[0];
}

// p loss + acc partials
__global__ void k_ploss(const float* __restrict__ p_logit, const int* __restrict__ p_tgt,
                        int Rows, float* __restrict__ acc) {
  __shared__ float sl[256], sc[256];
  const int i = blockIdx.x * blockDim.x + threadIdx.x;
  float loss = 0.f, corr = 0.f;
  if (i < Rows) {
    const float x = p_logit[i];
    const float t = (float)p_tgt[i];
    loss = fmaxf(x, 0.f) + log1pf(__expf(-fabsf(x))) - x * t;
    corr = (((x > 0.f) ? 1 : 0) == p_tgt[i]) ? 1.f : 0.f;
  }
  sl[threadIdx.x] = loss; sc[threadIdx.x] = corr;
  __syncthreads();
  for (int s = 128; s; s >>= 1) {
    if (threadIdx.x < s) { sl[threadIdx.x] += sl[threadIdx.x + s]; sc[threadIdx.x] += sc[threadIdx.x + s]; }
    __syncthreads();
  }
  if (threadIdx.x == 0) { atomicAdd(&acc[1], sl[0]); atomicAdd(&acc[3], sc[0]); }
}

// q loss + acc: wave per row (lse, argmax over V=780; q_lg ld NPV)
__global__ void k_qloss(const float* __restrict__ q_logits, const int* __restrict__ q_tgt,
                        int Rows, float* __restrict__ acc) {
  const int row = blockIdx.x * (blockDim.x >> 5) + (threadIdx.x >> 5);
  const int lane = threadIdx.x & 31;
  if (row >= Rows) return;
  const float* r = q_logits + (size_t)row * NPV;
  float mx = -3.0e38f; int ai = 0;
  for (int c = lane; c < VV; c += 32) {
    const float v = r[c];
    if (v > mx) { mx = v; ai = c; }
  }
  for (int off = 16; off; off >>= 1) {
    const float om = __shfl_down(mx, off, 32);
    const int   oi = __shfl_down(ai, off, 32);
    if (om > mx || (om == mx && oi < ai)) { mx = om; ai = oi; }
  }
  mx = __shfl(mx, 0, 32); ai = __shfl(ai, 0, 32);
  float se = 0.f;
  for (int c = lane; c < VV; c += 32) se += __expf(r[c] - mx);
  for (int off = 16; off; off >>= 1) se += __shfl_down(se, off, 32);
  if (lane == 0) {
    const int tgt = q_tgt[row];
    atomicAdd(&acc[0], mx + logf(se) - r[tgt]);
    atomicAdd(&acc[2], (ai == tgt) ? 1.f : 0.f);
  }
}

__global__ void k_final(const float* __restrict__ acc, float* __restrict__ out,
                        float nq, float np) {
  if (threadIdx.x == 0 && blockIdx.x == 0) {
    out[0] = acc[0] / (float)BB;   // q_loss
    out[1] = acc[1] / (float)BB;   // p_loss
    out[2] = acc[2] / nq;          // q_acc
    out[3] = acc[3] / np;          // p_acc
  }
}

// ---------------- host side ----------------
static inline char* ws_take(char*& p, size_t bytes) {
  char* r = p;
  p += (bytes + 255) & ~(size_t)255;
  return r;
}
static inline int ceil_div(long a, long b) { return (int)((a + b - 1) / b); }

extern "C" void kernel_launch(void* const* d_in, const int* in_sizes, int n_in,
                              void* d_out, int out_size, void* d_ws, size_t ws_size,
                              hipStream_t stream) {
  const float* tree_vec = (const float*)d_in[0];
  const float* emb = (const float*)d_in[1];
  const float* Wz = (const float*)d_in[2];  const float* bz = (const float*)d_in[3];
  const float* Wh = (const float*)d_in[4];  const float* bh = (const float*)d_in[5];
  const float* Wr = (const float*)d_in[6];  const float* Ur = (const float*)d_in[7];
  const float* br = (const float*)d_in[8];
  const float* Ww = (const float*)d_in[9];  const float* bw = (const float*)d_in[10];
  const float* Uw = (const float*)d_in[11]; const float* bu = (const float*)d_in[12];
  const float* Wo = (const float*)d_in[13]; const float* bo = (const float*)d_in[14];
  const float* Us = (const float*)d_in[15]; const float* bs = (const float*)d_in[16];
  const int* wid      = (const int*)d_in[17];
  const int* root_ids = (const int*)d_in[18];
  const int* edge_src = (const int*)d_in[19];
  const int* edge_dst = (const int*)d_in[20];
  const int* edge_pred = (const int*)d_in[21];
  const int* node_in   = (const int*)d_in[22];
  const int* step_eid  = (const int*)d_in[23];
  const int* step_v    = (const int*)d_in[24];
  const int* q_rows    = (const int*)d_in[25];
  const int* q_tgt     = (const int*)d_in[26];
  const int* p_tgt     = (const int*)d_in[27];

  const int P  = in_sizes[21] / NEDGES;
  const int Dn = in_sizes[22] / NNODES;
  const int nq = in_sizes[25];
  const int Mq = (nq + 63) & ~63;          // padded q-row count

  // ---- workspace carve-up ----
  char* wp = (char*)d_ws;
  _Float16* WzH   = (_Float16*)ws_take(wp, (size_t)KA   * NPH * 2);
  _Float16* WhH   = (_Float16*)ws_take(wp, (size_t)KA   * NPH * 2);
  _Float16* WrUrH = (_Float16*)ws_take(wp, (size_t)KA   * NPH * 2);
  _Float16* WwH   = (_Float16*)ws_take(wp, (size_t)KQIN * NPH * 2);
  _Float16* UwH   = (_Float16*)ws_take(wp, (size_t)KPIN * NPH * 2);
  _Float16* WoH   = (_Float16*)ws_take(wp, (size_t)KH2  * NPV * 2);
  float* bzP = (float*)ws_take(wp, NPH * 4);
  float* bhP = (float*)ws_take(wp, NPH * 4);
  float* brP = (float*)ws_take(wp, NPH * 4);
  float* bwP = (float*)ws_take(wp, NPH * 4);
  float* buP = (float*)ws_take(wp, NPH * 4);
  float* boP = (float*)ws_take(wp, NPV * 4);
  _Float16* x_h   = (_Float16*)ws_take(wp, (size_t)NNODES * HD * 2);
  float* m_st  = (float*)ws_take(wp, (size_t)(NEDGES + 1) * HD * 4);
  float* rm_st = (float*)ws_take(wp, (size_t)(NEDGES + 1) * HD * 4);
  float* h_all = (float*)ws_take(wp, (size_t)NPROWS * HD * 4);
  float* Sbuf  = (float*)ws_take(wp, (size_t)BB * NPH * 4);
  float* Zbuf  = (float*)ws_take(wp, (size_t)BB * NPH * 4);
  float* Hbuf  = (float*)ws_take(wp, (size_t)BB * NPH * 4);
  float* Mnew  = (float*)ws_take(wp, (size_t)BB * NPH * 4);
  float* Rbuf  = (float*)ws_take(wp, (size_t)BB * NPH * 4);
  _Float16* A1 = (_Float16*)ws_take(wp, (size_t)BB * KA * 2);
  _Float16* A2 = (_Float16*)ws_take(wp, (size_t)BB * KA * 2);
  _Float16* A3 = (_Float16*)ws_take(wp, (size_t)BB * KA * 2);
  _Float16* p_in  = (_Float16*)ws_take(wp, (size_t)NPROWS * KPIN * 2);
  _Float16* p_hid = (_Float16*)ws_take(wp, (size_t)NPROWS * NPH * 2);
  float*    p_lg  = (float*)ws_take(wp, (size_t)NPROWS * 4);
  _Float16* q_in  = (_Float16*)ws_take(wp, (size_t)Mq * KQIN * 2);
  _Float16* q_hid = (_Float16*)ws_take(wp, (size_t)Mq * NPH * 2);
  float*    q_lg  = (float*)ws_take(wp, (size_t)Mq * NPV * 4);
  float*    accp  = (float*)ws_take(wp, 4 * 4);

  // ---- zero state / pad regions (every call: deterministic) ----
  auto zero = [&](void* p, size_t bytes) {
    const size_t n32 = bytes / 4;
    int blocks = (int)((n32 + 255) / 256);
    if (blocks > 4096) blocks = 4096;
    k_zero_u32<<<blocks ? blocks : 1, 256, 0, stream>>>((unsigned*)p, n32);
  };
  zero(m_st,  (size_t)(NEDGES + 1) * HD * 4);
  zero(rm_st, (size_t)(NEDGES + 1) * HD * 4);
  zero(h_all, (size_t)NPROWS * HD * 4);
  zero(A1, (size_t)BB * KA * 2);
  zero(A2, (size_t)BB * KA * 2);
  zero(A3, (size_t)BB * KA * 2);
  zero(p_in, (size_t)NPROWS * KPIN * 2);
  zero(q_in, (size_t)Mq * KQIN * 2);
  zero(accp, 16);

  // ---- weights / biases to padded f16 / f32 ----
  auto cvt = [&](const float* s, _Float16* d, int K, int Nc, int KP, int NPc) {
    const size_t tot = (size_t)KP * NPc;
    k_cvt<<<ceil_div((long)tot, 256), 256, 0, stream>>>(s, d, K, Nc, KP, NPc);
  };
  cvt(Wz, WzH, 2 * HD, HD, KA, NPH);
  cvt(Wh, WhH, 2 * HD, HD, KA, NPH);
  cvt(Wr, WrUrH, HD, HD, HD, NPH);                         // rows 0..449
  cvt(Ur, WrUrH + (size_t)HD * NPH, HD, HD, KA - HD, NPH); // rows 450..927 (pad zeroed)
  cvt(Ww, WwH, HD + LD, HD, KQIN, NPH);
  cvt(Uw, UwH, 2 * HD + LD, HD, KPIN, NPH);
  cvt(Wo, WoH, HD, VV, KH2, NPV);
  k_cvt_bias<<<2, 256, 0, stream>>>(bz, bzP, HD, NPH);
  k_cvt_bias<<<2, 256, 0, stream>>>(bh, bhP, HD, NPH);
  k_cvt_bias<<<2, 256, 0, stream>>>(br, brP, HD, NPH);
  k_cvt_bias<<<2, 256, 0, stream>>>(bw, bwP, HD, NPH);
  k_cvt_bias<<<2, 256, 0, stream>>>(bu, buP, HD, NPH);
  k_cvt_bias<<<4, 256, 0, stream>>>(bo, boP, VV, NPV);

  k_gather_emb<<<ceil_div((long)NNODES * HD, 256), 256, 0, stream>>>(emb, wid, x_h);

  // ---- sequential DFS scan ----
  const int ebh = ceil_div((long)BB * HD, 256);
  const dim3 gH(NPH / TBN, BB / TBM);        // 4 x 8 blocks for the recurrent GEMMs
  for (int t = 0; t < TT; ++t) {
    const int* eid = step_eid + (size_t)t * BB;
    const int* sv  = step_v   + (size_t)t * BB;
    k_step_build<<<ebh, 256, 0, stream>>>(eid, edge_src, edge_pred, P, m_st, rm_st, x_h,
                                          Sbuf, A1, A2);
    k_gemm<1, 0><<<gH, 256, 0, stream>>>(A1, WzH, bzP, Zbuf, NPH, KA);   // sigmoid
    k_gemm<2, 0><<<gH, 256, 0, stream>>>(A2, WhH, bhP, Hbuf, NPH, KA);   // tanh
    k_mnew<<<ebh, 256, 0, stream>>>(eid, edge_dst, Zbuf, Hbuf, Sbuf, x_h, Mnew, A3);
    k_gemm<1, 0><<<gH, 256, 0, stream>>>(A3, WrUrH, brP, Rbuf, NPH, KA); // sigmoid
    k_scatter<<<ebh, 256, 0, stream>>>(eid, Mnew, Rbuf, m_st, rm_st);
    k_pull<<<ebh, 256, 0, stream>>>(sv, node_in, Dn, m_st,
                                    h_all + (size_t)(t + 1) * BB * HD);
  }

  // ---- output heads ----
  k_build_pin<<<ceil_div((long)NPROWS * (2 * HD + LD), 256), 256, 0, stream>>>(
      x_h, h_all, tree_vec, root_ids, step_v, p_in);
  k_build_qin<<<ceil_div((long)nq * (HD + LD), 256), 256, 0, stream>>>(
      h_all, tree_vec, q_rows, q_in, nq);

  const dim3 gp(NPH / TBN, NPROWS / TBM);
  k_gemm<3, 1><<<gp, 256, 0, stream>>>(p_in, UwH, buP, p_hid, NPH, KPIN); // relu -> f16
  k_plogit<<<ceil_div(NPROWS, 8), 256, 0, stream>>>(p_hid, Us, bs, p_lg, NPROWS);
  k_ploss<<<ceil_div(NPROWS, 256), 256, 0, stream>>>(p_lg, p_tgt, NPROWS, accp);

  const dim3 gq1(NPH / TBN, Mq / TBM);
  k_gemm<3, 1><<<gq1, 256, 0, stream>>>(q_in, WwH, bwP, q_hid, NPH, KQIN); // relu -> f16
  const dim3 gq2(NPV / TBN, Mq / TBM);
  k_gemm<0, 0><<<gq2, 256, 0, stream>>>(q_hid, WoH, boP, q_lg, NPV, KH2);  // logits f32
  k_qloss<<<ceil_div(nq, 8), 256, 0, stream>>>(q_lg, q_tgt, nq, accp);

  k_final<<<1, 32, 0, stream>>>(accp, (float*)d_out, (float)nq, (float)NPROWS);
}
